// GeomBiasTransformerEncoderLayer_11510512353787
// MI455X (gfx1250) — matrix operand order repaired and checked by reading, not compiled, not run on hardware
//
#include <hip/hip_runtime.h>
#include <math.h>

// ---------------------------------------------------------------------------
// CDNA5 (gfx1250) transformer encoder layer: bf16 WMMA everywhere.
// ---------------------------------------------------------------------------

typedef __bf16 bf16;
typedef __attribute__((ext_vector_type(16))) __bf16 v16bf;
typedef __attribute__((ext_vector_type(8)))  __bf16 v8bf;
typedef __attribute__((ext_vector_type(8)))  float  v8f;

union AF { v16bf v; v8bf h[2]; };

static constexpr int kB  = 4;
static constexpr int kL  = 2048;
static constexpr int kD  = 512;
static constexpr int kH  = 8;
static constexpr int kDh = 64;
static constexpr int kF  = 2048;
static constexpr int kM  = kB * kL;   // 8192 rows

#define LN_EPS 1e-5f

__device__ inline v8f vzero8() {
  v8f z;
#pragma unroll
  for (int i = 0; i < 8; ++i) z[i] = 0.0f;
  return z;
}

__device__ inline v8f wmma_bf16(const AF& a, const AF& b, v8f c) {
  // D = A(16x32 bf16) * B(32x16 bf16) + C(16x16 f32)
  return __builtin_amdgcn_wmma_f32_16x16x32_bf16(
      /*neg_a=*/false, a.v, /*neg_b=*/false, b.v,
      /*c_mod=*/(short)0, c, /*reuse_a=*/false, /*reuse_b=*/false);
}

// ---------------------------------------------------------------------------
// Weight transpose + f32->bf16 convert: w[K][N] -> wT[N][K]
// ---------------------------------------------------------------------------
__global__ void wtrans_kernel(const float* __restrict__ w, bf16* __restrict__ wT,
                              int K, int N) {
  int idx = blockIdx.x * blockDim.x + threadIdx.x;
  if (idx >= K * N) return;
  int kk = idx / N;
  int n  = idx - kk * N;
  wT[(size_t)n * K + kk] = (bf16)w[idx];
}

// ---------------------------------------------------------------------------
// V head transpose: v[b*L + l][h*64 + d] (bf16) -> vT[((b*H+h)*64+d)*L + l]
// ---------------------------------------------------------------------------
__global__ void vtrans_kernel(const bf16* __restrict__ v, bf16* __restrict__ vT) {
  int idx = blockIdx.x * blockDim.x + threadIdx.x;   // over B*H*Dh*L
  if (idx >= kB * kH * kDh * kL) return;
  int l = idx & (kL - 1);
  int t = idx >> 11;          // / L
  int d = t & (kDh - 1);
  t >>= 6;                    // / Dh
  int h = t & (kH - 1);
  int b = t >> 3;             // / H
  vT[idx] = v[((size_t)(b * kL + l)) * kD + h * kDh + d];
}

// ---------------------------------------------------------------------------
// Row LayerNorm (D=512) with bf16 output. One block (128 threads) per row.
// ---------------------------------------------------------------------------
__global__ void __launch_bounds__(128) layernorm_bf16_kernel(
    const float* __restrict__ x, const float* __restrict__ g,
    const float* __restrict__ b, bf16* __restrict__ out) {
  __shared__ float red[128];
  const int row = blockIdx.x;
  const int tid = threadIdx.x;
  const float* xr = x + (size_t)row * kD;
  float v[4];
  float s = 0.f;
#pragma unroll
  for (int i = 0; i < 4; ++i) { v[i] = xr[tid + i * 128]; s += v[i]; }
  red[tid] = s;
  __syncthreads();
  for (int off = 64; off > 0; off >>= 1) {
    if (tid < off) red[tid] += red[tid + off];
    __syncthreads();
  }
  const float mu = red[0] * (1.0f / kD);
  __syncthreads();
  float sq = 0.f;
#pragma unroll
  for (int i = 0; i < 4; ++i) { float d = v[i] - mu; sq += d * d; }
  red[tid] = sq;
  __syncthreads();
  for (int off = 64; off > 0; off >>= 1) {
    if (tid < off) red[tid] += red[tid + off];
    __syncthreads();
  }
  const float inv = rsqrtf(red[0] * (1.0f / kD) + LN_EPS);
#pragma unroll
  for (int i = 0; i < 4; ++i) {
    int c = tid + i * 128;
    out[(size_t)row * kD + c] = (bf16)((v[i] - mu) * inv * g[c] + b[c]);
  }
}

// ---------------------------------------------------------------------------
// Generic bf16 WMMA GEMM: C[M][N] = A[M][K] @ Bt[N][K]^T + bias (+ epilogue)
// Block = 128 threads = 4 waves. Block tile 64(M) x 64(N); wave tile 16 x 64.
// MODE 0: out bf16        MODE 1: exact GELU -> bf16    MODE 2: +resid -> f32
// ---------------------------------------------------------------------------
template <int MODE>
__global__ void __launch_bounds__(128) gemm_bf16_kernel(
    const bf16* __restrict__ A, const bf16* __restrict__ Bt,
    const float* __restrict__ bias, const float* __restrict__ resid,
    float* __restrict__ outF, bf16* __restrict__ outB,
    int M, int N, int K) {
  const int lane = threadIdx.x & 31;
  const int wv   = threadIdx.x >> 5;
  const int lm   = lane & 15;     // M (A) or N (B,C) within tile
  const int kh   = lane >> 4;     // half-wave selector
  const int m0 = blockIdx.y * 64 + wv * 16;
  const int n0 = blockIdx.x * 64;

  v8f acc[4];
#pragma unroll
  for (int j = 0; j < 4; ++j) acc[j] = vzero8();

  for (int k0 = 0; k0 < K; k0 += 32) {
    // A fragment: lane holds row (m0+lm); K chunks [kh*8, kh*8+8) and +16.
    AF af;
    const bf16* ap = A + (size_t)(m0 + lm) * K + k0 + kh * 8;
    af.h[0] = *(const v8bf*)ap;
    af.h[1] = *(const v8bf*)(ap + 16);
#pragma unroll
    for (int j = 0; j < 4; ++j) {
      // B fragment: lane holds col (n0+j*16+lm); K chunks [kh*8, kh*8+8) and +16.
      AF bfr;
      const bf16* bp = Bt + (size_t)(n0 + j * 16 + lm) * K + k0 + kh * 8;
      bfr.h[0] = *(const v8bf*)bp;
      bfr.h[1] = *(const v8bf*)(bp + 16);
      acc[j] = wmma_bf16(af, bfr, acc[j]);
    }
  }

#pragma unroll
  for (int j = 0; j < 4; ++j) {
    const int n = n0 + j * 16 + lm;
    const float bn = bias[n];
#pragma unroll
    for (int r = 0; r < 8; ++r) {
      const int m = m0 + kh * 8 + r;     // C layout: VGPR r -> M = 8*kh + r
      float val = acc[j][r] + bn;
      const size_t oi = (size_t)m * N + n;
      if (MODE == 0) {
        outB[oi] = (bf16)val;
      } else if (MODE == 1) {
        outB[oi] = (bf16)(0.5f * val * (1.0f + erff(val * 0.70710678118f)));
      } else {
        outF[oi] = resid[oi] + val;
      }
    }
  }
}

// ---------------------------------------------------------------------------
// Flash-style attention. Grid (L/64, H, B), 4 waves/block; wave owns 16 q rows.
// scores = (Q Kt) * 0.125 + bias; online softmax; O += P V.
// ---------------------------------------------------------------------------
__global__ void __launch_bounds__(128) attn_kernel(
    const bf16* __restrict__ q, const bf16* __restrict__ k,
    const bf16* __restrict__ vT, const float* __restrict__ bias,
    bf16* __restrict__ o) {
  __shared__ __align__(64) bf16 plds[4][16][32];   // per-wave P tile stash
  const int lane = threadIdx.x & 31;
  const int wv   = threadIdx.x >> 5;
  const int lm   = lane & 15;
  const int kh   = lane >> 4;
  const int b  = blockIdx.z;
  const int h  = blockIdx.y;
  const int qt = blockIdx.x * 64 + wv * 16;
  const float scale = 0.125f;   // Dh^-0.5

  // Q fragments for K-dim = Dh = 64 (two k-steps of 32), kept in registers.
  AF qf[2];
#pragma unroll
  for (int kk = 0; kk < 2; ++kk) {
    const bf16* qp = q + (size_t)(b * kL + qt + lm) * kD + h * kDh + kk * 32 + kh * 8;
    qf[kk].h[0] = *(const v8bf*)qp;
    qf[kk].h[1] = *(const v8bf*)(qp + 16);
  }

  v8f oacc[4];
#pragma unroll
  for (int j = 0; j < 4; ++j) oacc[j] = vzero8();
  float mrow[8], lrow[8];
#pragma unroll
  for (int r = 0; r < 8; ++r) { mrow[r] = -1e30f; lrow[r] = 0.0f; }

#pragma unroll 1
  for (int kt = 0; kt < kL; kt += 32) {
    // ---- scores: 16 q-rows x 32 keys (two 16-col WMMA tiles) ----
    v8f s[2];
#pragma unroll
    for (int j = 0; j < 2; ++j) {
      s[j] = vzero8();
#pragma unroll
      for (int kk = 0; kk < 2; ++kk) {
        AF bfr;   // Kt fragment: lane = key (n), Dh chunks [kh*8, kh*8+8) and +16
        const bf16* kp = k + (size_t)(b * kL + kt + j * 16 + lm) * kD
                           + h * kDh + kk * 32 + kh * 8;
        bfr.h[0] = *(const v8bf*)kp;
        bfr.h[1] = *(const v8bf*)(kp + 16);
        s[j] = wmma_bf16(qf[kk], bfr, s[j]);
      }
    }
    // ---- bias add + online softmax (row = 8*kh + r, col = j*16 + lm) ----
    float p0s[8], p1s[8];
#pragma unroll
    for (int r = 0; r < 8; ++r) {
      const int qrow = qt + kh * 8 + r;
      const size_t brow = ((size_t)h * kL + qrow) * kL + kt;
      float v0 = s[0][r] * scale + bias[brow + lm];
      float v1 = s[1][r] * scale + bias[brow + 16 + lm];
      float tm = fmaxf(v0, v1);
#pragma unroll
      for (int msk = 1; msk < 16; msk <<= 1)
        tm = fmaxf(tm, __shfl_xor(tm, msk, 32));
      const float mnew = fmaxf(mrow[r], tm);
      const float fac  = expf(mrow[r] - mnew);
      mrow[r] = mnew;
      const float p0 = expf(v0 - mnew);
      const float p1 = expf(v1 - mnew);
      float rs = p0 + p1;
#pragma unroll
      for (int msk = 1; msk < 16; msk <<= 1)
        rs += __shfl_xor(rs, msk, 32);
      lrow[r] = lrow[r] * fac + rs;
#pragma unroll
      for (int j2 = 0; j2 < 4; ++j2) oacc[j2][r] *= fac;
      p0s[r] = p0; p1s[r] = p1;
    }
    // ---- C-layout -> A-layout via per-wave LDS bounce ----
#pragma unroll
    for (int r = 0; r < 8; ++r) {
      plds[wv][kh * 8 + r][lm]      = (bf16)p0s[r];
      plds[wv][kh * 8 + r][16 + lm] = (bf16)p1s[r];
    }
    asm volatile("s_wait_dscnt 0" ::: "memory");   // wave-local DS ordering
    AF pf;
    pf.h[0] = *(const v8bf*)&plds[wv][lm][kh * 8];
    pf.h[1] = *(const v8bf*)&plds[wv][lm][16 + kh * 8];
    asm volatile("" ::: "memory");
    // ---- O += P(16x32) @ V(32x64): V pre-transposed to [b][h][dh][l] ----
#pragma unroll
    for (int j2 = 0; j2 < 4; ++j2) {
      AF vf;
      const bf16* vp = vT + ((size_t)(b * kH + h) * kDh + j2 * 16 + lm) * kL
                          + kt + kh * 8;
      vf.h[0] = *(const v8bf*)vp;
      vf.h[1] = *(const v8bf*)(vp + 16);
      oacc[j2] = wmma_bf16(pf, vf, oacc[j2]);
    }
  }

  // ---- normalize and write O (bf16, [B*L][D] layout for the wo GEMM) ----
#pragma unroll
  for (int j2 = 0; j2 < 4; ++j2) {
#pragma unroll
    for (int r = 0; r < 8; ++r) {
      const int row = qt + kh * 8 + r;
      const float val = oacc[j2][r] / lrow[r];
      o[(size_t)(b * kL + row) * kD + h * kDh + j2 * 16 + lm] = (bf16)val;
    }
  }
}

// ---------------------------------------------------------------------------
// Host launch
// ---------------------------------------------------------------------------
extern "C" void kernel_launch(void* const* d_in, const int* in_sizes, int n_in,
                              void* d_out, int out_size, void* d_ws, size_t ws_size,
                              hipStream_t stream) {
  (void)in_sizes; (void)n_in; (void)out_size; (void)ws_size;
  const float* x   = (const float*)d_in[0];
  const float* ab  = (const float*)d_in[1];
  const float* l1w = (const float*)d_in[2];
  const float* l1b = (const float*)d_in[3];
  const float* l2w = (const float*)d_in[4];
  const float* l2b = (const float*)d_in[5];
  const float* wq  = (const float*)d_in[6];
  const float* bq  = (const float*)d_in[7];
  const float* wk  = (const float*)d_in[8];
  const float* bk  = (const float*)d_in[9];
  const float* wvw = (const float*)d_in[10];
  const float* bv  = (const float*)d_in[11];
  const float* wo  = (const float*)d_in[12];
  const float* bo  = (const float*)d_in[13];
  const float* w1  = (const float*)d_in[14];
  const float* b1  = (const float*)d_in[15];
  const float* w2  = (const float*)d_in[16];
  const float* b2  = (const float*)d_in[17];

  char* ws = (char*)d_ws;
  size_t off = 0;
  auto take = [&](size_t bytes) {
    char* p = ws + off;
    off += (bytes + 255) & ~(size_t)255;
    return p;
  };

  bf16* wqT = (bf16*)take((size_t)kD * kD * 2);
  bf16* wkT = (bf16*)take((size_t)kD * kD * 2);
  bf16* wvT = (bf16*)take((size_t)kD * kD * 2);
  bf16* woT = (bf16*)take((size_t)kD * kD * 2);
  bf16* w1T = (bf16*)take((size_t)kD * kF * 2);
  bf16* w2T = (bf16*)take((size_t)kF * kD * 2);
  bf16* h1  = (bf16*)take((size_t)kM * kD * 2);
  bf16* qb  = (bf16*)take((size_t)kM * kD * 2);
  bf16* kb  = (bf16*)take((size_t)kM * kD * 2);
  bf16* vb  = (bf16*)take((size_t)kM * kD * 2);
  bf16* vTb = (bf16*)take((size_t)kM * kD * 2);
  bf16* ob  = (bf16*)take((size_t)kM * kD * 2);
  float* x1 = (float*)take((size_t)kM * kD * 4);
  bf16* h2  = (bf16*)take((size_t)kM * kD * 2);
  bf16* ff  = (bf16*)take((size_t)kM * kF * 2);

  // Weight transposes (f32 -> bf16, N-major so GEMM B-fragments are contiguous)
  wtrans_kernel<<<(kD * kD + 255) / 256, 256, 0, stream>>>(wq, wqT, kD, kD);
  wtrans_kernel<<<(kD * kD + 255) / 256, 256, 0, stream>>>(wk, wkT, kD, kD);
  wtrans_kernel<<<(kD * kD + 255) / 256, 256, 0, stream>>>(wvw, wvT, kD, kD);
  wtrans_kernel<<<(kD * kD + 255) / 256, 256, 0, stream>>>(wo, woT, kD, kD);
  wtrans_kernel<<<(kD * kF + 255) / 256, 256, 0, stream>>>(w1, w1T, kD, kF);
  wtrans_kernel<<<(kF * kD + 255) / 256, 256, 0, stream>>>(w2, w2T, kF, kD);

  // ---- attention block ----
  layernorm_bf16_kernel<<<kM, 128, 0, stream>>>(x, l1w, l1b, h1);
  dim3 gP(kD / 64, kM / 64);
  gemm_bf16_kernel<0><<<gP, 128, 0, stream>>>(h1, wqT, bq, nullptr, nullptr, qb, kM, kD, kD);
  gemm_bf16_kernel<0><<<gP, 128, 0, stream>>>(h1, wkT, bk, nullptr, nullptr, kb, kM, kD, kD);
  gemm_bf16_kernel<0><<<gP, 128, 0, stream>>>(h1, wvT, bv, nullptr, nullptr, vb, kM, kD, kD);
  vtrans_kernel<<<(kM * kD + 255) / 256, 256, 0, stream>>>(vb, vTb);
  attn_kernel<<<dim3(kL / 64, kH, kB), 128, 0, stream>>>(qb, kb, vTb, ab, ob);
  gemm_bf16_kernel<2><<<gP, 128, 0, stream>>>(ob, woT, bo, x, x1, nullptr, kM, kD, kD);

  // ---- feed-forward block ----
  layernorm_bf16_kernel<<<kM, 128, 0, stream>>>(x1, l2w, l2b, h2);
  dim3 gF(kF / 64, kM / 64);
  gemm_bf16_kernel<1><<<gF, 128, 0, stream>>>(h2, w1T, b1, nullptr, nullptr, ff, kM, kF, kD);
  gemm_bf16_kernel<2><<<gP, 128, 0, stream>>>(ff, w2T, b2, x1, (float*)d_out, nullptr, kM, kD, kF);
}
